// GNNML3_64991445123411
// MI455X (gfx1250) — matrix operand order repaired and checked by reading, not compiled, not run on hardware
//
#include <hip/hip_runtime.h>
#include <cstdint>
#include <cstddef>

typedef __attribute__((ext_vector_type(16))) _Float16 v16h;
typedef __attribute__((ext_vector_type(8)))  float    v8f;

#define NOUTC 24
#define NINC  48
#define ZC    192   // z columns (8*24)
#define GC    64    // raw gate columns (A:0..23, pad, B:32..55, pad)
#define KPAD  64    // padded reduction dim
#define WCOLS 256   // padded output cols of fused weight panel

// ---------------- build fused transposed f16 weight panel WT[256][64] ----------------
// col c: [0,192): W[k=c/24][i][j=c%24]; [192,216): A[i][c-192]; [224,248): B[i][c-224]; else 0
__global__ void pack_w_k(const float* __restrict__ W, const float* __restrict__ A,
                         const float* __restrict__ B, _Float16* __restrict__ WT, int din) {
  int t = blockIdx.x * blockDim.x + threadIdx.x;
  if (t >= WCOLS * KPAD) return;
  int c = t >> 6, i = t & 63;
  float v = 0.f;
  if (i < din) {
    if (c < ZC)                    v = W[((c / NOUTC) * din + i) * NOUTC + (c % NOUTC)];
    else if (c < ZC + NOUTC)       v = A[i * NOUTC + (c - ZC)];
    else if (c >= 224 && c < 224 + NOUTC) v = B[i * NOUTC + (c - 224)];
  }
  WT[t] = (_Float16)v;
}

// ---------------- f32 -> f16 input convert (layer 1, din==KPAD==64) ----------------
__global__ void cvt_x_k(const float* __restrict__ x, _Float16* __restrict__ xh, int tot) {
  int t = blockIdx.x * blockDim.x + threadIdx.x;
  if (t < tot) xh[t] = (_Float16)x[t];
}

// ---------------- fused matmul: [N,64]f16 x [64,256]f16 -> z[N,192], gate raw[N,64] ----------------
__global__ __launch_bounds__(256) void mm_wmma_k(const _Float16* __restrict__ Xh,
                                                 const _Float16* __restrict__ WT,
                                                 float* __restrict__ zb,
                                                 float* __restrict__ gb, int nrows) {
  const int lane = threadIdx.x & 31;
  const int wave = threadIdx.x >> 5;
  const int rt   = blockIdx.x * 8 + wave;
  const int row0 = rt * 16;
  if (row0 >= nrows) return;                 // wave-uniform: EXEC stays all-ones
  const int col0 = blockIdx.y * 16;
  const int m  = lane & 15;
  const int kh = lane >> 4;                  // 0: K 0..7 / 16..23 ; 1: K 8..15 / 24..31
  const _Float16* xa = Xh + (size_t)(row0 + m) * KPAD + kh * 8;
  const _Float16* wb = WT + (size_t)(col0 + m) * KPAD + kh * 8;
  v8f acc = {};
#pragma unroll
  for (int kb = 0; kb < KPAD; kb += 32) {
    v16h a, b;
#pragma unroll
    for (int i = 0; i < 8; ++i) { a[i] = xa[kb + i]; a[8 + i] = xa[kb + 16 + i]; }
#pragma unroll
    for (int i = 0; i < 8; ++i) { b[i] = wb[kb + i]; b[8 + i] = wb[kb + 16 + i]; }
    acc = __builtin_amdgcn_wmma_f32_16x16x32_f16(false, a, false, b,
                                                 (short)0, acc, false, false);
  }
  const int rb = row0 + kh * 8;
  const int cw = col0 + m;
  if (col0 < ZC) {                           // uniform per block (col0 multiple of 16)
#pragma unroll
    for (int j = 0; j < 8; ++j) zb[(size_t)(rb + j) * ZC + cw] = acc[j];
  } else {
#pragma unroll
    for (int j = 0; j < 8; ++j) gb[(size_t)(rb + j) * GC + (cw - ZC)] = acc[j];
  }
}

// ---------------- per-edge: acc[dst] += sum_k ea[e,k] * z[src][k*24+c] ----------------
__global__ void edge_scatter_k(const int* __restrict__ src, const int* __restrict__ dst,
                               const float* __restrict__ ea, const float* __restrict__ zb,
                               float* __restrict__ acc, int E) {
  int t = blockIdx.x * blockDim.x + threadIdx.x;
  int e = t >> 5, lane = t & 31;
  if (e >= E || lane >= NOUTC) return;
  int s = src[e], d = dst[e];
  const float* zp = zb + (size_t)s * ZC + lane;
  const float* ep = ea + (size_t)e * 8;
  float sum = 0.f;
#pragma unroll
  for (int k = 0; k < 8; ++k) sum += ep[k] * zp[k * NOUTC];
  atomicAdd(acc + (size_t)d * NOUTC + lane, sum);
}

// ---------------- h[:,0:24]=relu(acc+b); h[:,24:48]=relu(gA+a)*tanh(gB+c) ----------------
__global__ void assemble_h_k(const float* __restrict__ acc, const float* __restrict__ bias,
                             const float* __restrict__ gbuf, const float* __restrict__ ga,
                             const float* __restrict__ gc, float* __restrict__ h, int nrows) {
  int t = blockIdx.x * blockDim.x + threadIdx.x;
  int n = t >> 5, j = t & 31;
  if (n >= nrows || j >= NOUTC) return;
  float conv = fmaxf(acc[(size_t)n * NOUTC + j] + bias[j], 0.f);
  float u = fmaxf(gbuf[(size_t)n * GC + j] + ga[j], 0.f);
  float w = tanhf(gbuf[(size_t)n * GC + 32 + j] + gc[j]);
  h[(size_t)n * NINC + j] = conv;
  h[(size_t)n * NINC + NOUTC + j] = u * w;
}

// ---------------- BN stats: per-column sum / sumsq ----------------
__global__ void stats_k(const float* __restrict__ h, float* __restrict__ stats, int nrows) {
  __shared__ float ssum[256], ssq[256];
  int col = blockIdx.x;
  float s = 0.f, q = 0.f;
  for (int n = blockIdx.y * blockDim.x + threadIdx.x; n < nrows; n += blockDim.x * gridDim.y) {
    float v = h[(size_t)n * NINC + col];
    s += v; q += v * v;
  }
  ssum[threadIdx.x] = s; ssq[threadIdx.x] = q;
  __syncthreads();
  for (int st = 128; st > 0; st >>= 1) {
    if (threadIdx.x < st) {
      ssum[threadIdx.x] += ssum[threadIdx.x + st];
      ssq[threadIdx.x]  += ssq[threadIdx.x + st];
    }
    __syncthreads();
  }
  if (threadIdx.x == 0) {
    atomicAdd(&stats[col], ssum[0]);
    atomicAdd(&stats[NINC + col], ssq[0]);
  }
}

// ---------------- BN apply; write f32 h in-place and padded f16 next-layer input ----------------
__global__ void bn_apply_k(float* __restrict__ h, const float* __restrict__ stats,
                           const float* __restrict__ g, const float* __restrict__ be,
                           _Float16* __restrict__ xh, int nrows) {
  int t = blockIdx.x * blockDim.x + threadIdx.x;
  int n = t >> 6, c = t & 63;
  if (n >= nrows) return;
  if (c >= NINC) { xh[(size_t)n * KPAD + c] = (_Float16)0.f; return; }
  float inv = 1.0f / (float)nrows;
  float mean = stats[c] * inv;
  float var  = stats[NINC + c] * inv - mean * mean;
  float scl  = g[c] * rsqrtf(var + 1e-5f);
  float v = (h[(size_t)n * NINC + c] - mean) * scl + be[c];
  h[(size_t)n * NINC + c] = v;
  xh[(size_t)n * KPAD + c] = (_Float16)v;
}

// ---------------- mean pool accumulate ----------------
__global__ void pool_k(const float* __restrict__ h, const int* __restrict__ batch,
                       float* __restrict__ pooled, float* __restrict__ cnt, int nrows) {
  int t = blockIdx.x * blockDim.x + threadIdx.x;
  int n = t >> 6, c = t & 63;
  if (n >= nrows || c >= NINC) return;
  int g = batch[n];
  atomicAdd(&pooled[(size_t)g * NINC + c], h[(size_t)n * NINC + c]);
  if (c == 0) atomicAdd(&cnt[g], 1.0f);
}

// ---------------- final MLP: one wave32 block per graph ----------------
__global__ void mlp_k(const float* __restrict__ pooled, const float* __restrict__ cnt,
                      const float* __restrict__ F1, const float* __restrict__ f1,
                      const float* __restrict__ F2, const float* __restrict__ f2,
                      float* __restrict__ out) {
  int g = blockIdx.x;
  int lane = threadIdx.x;           // 32 lanes -> 32 hidden units
  float inv = 1.0f / fmaxf(cnt[g], 1.0f);
  float acc = f1[lane];
  for (int c = 0; c < NINC; ++c)
    acc += pooled[(size_t)g * NINC + c] * inv * F1[c * 32 + lane];
  acc = fmaxf(acc, 0.f);
  float v = acc * F2[lane];
  for (int off = 16; off > 0; off >>= 1) v += __shfl_down(v, off, 32);
  if (lane == 0) out[g] = v + f2[0];
}

extern "C" void kernel_launch(void* const* d_in, const int* in_sizes, int n_in,
                              void* d_out, int out_size, void* d_ws, size_t ws_size,
                              hipStream_t stream) {
  (void)n_in; (void)ws_size;
  const float* x     = (const float*)d_in[0];
  const int*   ei    = (const int*)d_in[1];
  const float* ea    = (const float*)d_in[2];
  const int*   batch = (const int*)d_in[3];
  const int N = in_sizes[0] / 64;
  const int E = in_sizes[1] / 2;
  const int G = out_size;
  const int* srcp = ei;
  const int* dstp = ei + E;

  char* ws = (char*)d_ws;
  size_t off = 0;
  auto alloc = [&](size_t bytes) -> void* {
    void* p = ws + off;
    off = (off + bytes + 255) & ~(size_t)255;
    return p;
  };
  float*     zb     = (float*)alloc((size_t)N * ZC * 4);
  float*     gbuf   = (float*)alloc((size_t)N * GC * 4);
  float*     accb   = (float*)alloc((size_t)N * NOUTC * 4);
  float*     hb     = (float*)alloc((size_t)N * NINC * 4);
  _Float16*  xh     = (_Float16*)alloc((size_t)N * KPAD * 2);
  _Float16*  WT     = (_Float16*)alloc((size_t)WCOLS * KPAD * 2);
  float*     stats  = (float*)alloc(2 * NINC * 4);
  float*     pooled = (float*)alloc((size_t)G * NINC * 4 + (size_t)G * 4);
  float*     cnt    = pooled + (size_t)G * NINC;

  const float* Wl[3]  = {(const float*)d_in[4],  (const float*)d_in[12], (const float*)d_in[20]};
  const float* bl[3]  = {(const float*)d_in[5],  (const float*)d_in[13], (const float*)d_in[21]};
  const float* Al[3]  = {(const float*)d_in[6],  (const float*)d_in[14], (const float*)d_in[22]};
  const float* al[3]  = {(const float*)d_in[7],  (const float*)d_in[15], (const float*)d_in[23]};
  const float* Bl[3]  = {(const float*)d_in[8],  (const float*)d_in[16], (const float*)d_in[24]};
  const float* cl[3]  = {(const float*)d_in[9],  (const float*)d_in[17], (const float*)d_in[25]};
  const float* gl[3]  = {(const float*)d_in[10], (const float*)d_in[18], (const float*)d_in[26]};
  const float* bel[3] = {(const float*)d_in[11], (const float*)d_in[19], (const float*)d_in[27]};
  const float* F1 = (const float*)d_in[28];
  const float* f1 = (const float*)d_in[29];
  const float* F2 = (const float*)d_in[30];
  const float* f2 = (const float*)d_in[31];
  const int dins[3] = {64, 48, 48};

  // layer-1 f16 input (din==64 fills all padded columns)
  {
    int tot = N * KPAD;
    cvt_x_k<<<(tot + 255) / 256, 256, 0, stream>>>(x, xh, tot);
  }
  const int rowTiles = (N + 15) / 16;
  dim3 mmGrid((rowTiles + 7) / 8, WCOLS / 16);

  for (int l = 0; l < 3; ++l) {
    pack_w_k<<<(WCOLS * KPAD + 255) / 256, 256, 0, stream>>>(Wl[l], Al[l], Bl[l], WT, dins[l]);
    (void)hipMemsetAsync(accb, 0, (size_t)N * NOUTC * 4, stream);
    (void)hipMemsetAsync(stats, 0, 2 * NINC * 4, stream);

    mm_wmma_k<<<mmGrid, 256, 0, stream>>>(xh, WT, zb, gbuf, N);

    {
      long long tot = (long long)E * 32;
      edge_scatter_k<<<(unsigned)((tot + 255) / 256), 256, 0, stream>>>(srcp, dstp, ea, zb, accb, E);
    }
    {
      long long tot = (long long)N * 32;
      assemble_h_k<<<(unsigned)((tot + 255) / 256), 256, 0, stream>>>(accb, bl[l], gbuf, al[l], cl[l], hb, N);
    }
    stats_k<<<dim3(NINC, 16), 256, 0, stream>>>(hb, stats, N);
    {
      long long tot = (long long)N * KPAD;
      bn_apply_k<<<(unsigned)((tot + 255) / 256), 256, 0, stream>>>(hb, stats, gl[l], bel[l], xh, N);
    }
  }

  (void)hipMemsetAsync(pooled, 0, (size_t)G * NINC * 4 + (size_t)G * 4, stream);
  {
    long long tot = (long long)N * 64;
    pool_k<<<(unsigned)((tot + 255) / 256), 256, 0, stream>>>(hb, batch, pooled, cnt, N);
  }
  mlp_k<<<G, 32, 0, stream>>>(pooled, cnt, F1, f1, F2, f2, (float*)d_out);
}
